// IdxSelMLP_29403346108725
// MI455X (gfx1250) — matrix operand order repaired and checked by reading, compiled, not run on hardware
//
#include <hip/hip_runtime.h>
#include <hip/hip_bf16.h>

// ---------------- problem constants ----------------
#define BATCH   2048
#define L0      1024
#define L1      2048
#define L2      1024
#define OPTIONS 4

#define BM   64
#define BN   128
#define BK   32
#define KPAD 40              // LDS row stride in bf16 elems (80B = 20 banks: conflict-free frag reads)
#define MT   36              // max padded 64-row tiles: 2048 + 4*63 -> 2304 rows -> 36 tiles
#define PB   (MT * BM)       // 2304 padded rows

typedef __attribute__((ext_vector_type(16))) __bf16 v16bf;
typedef __attribute__((ext_vector_type(8)))  __bf16 v8bf;
typedef __attribute__((ext_vector_type(2)))  __bf16 v2bf;
typedef __attribute__((ext_vector_type(8)))  float  v8f;

union Frag { v16bf v; v8bf h[2]; };

// ---------------- prep: deterministic group-by-option ----------------
__global__ __launch_bounds__(256) void idxmlp_prep(const int* __restrict__ idx,
                                                   int* __restrict__ perm,
                                                   int* __restrict__ tileOpt) {
    __shared__ unsigned char sIdx[BATCH];
    __shared__ int sCount[OPTIONS];
    __shared__ int sOff[OPTIONS];
    int t = threadIdx.x;
    for (int r = t; r < BATCH; r += 256) sIdx[r] = (unsigned char)idx[r];
    for (int s = t; s < PB; s += 256) perm[s] = -1;
    __syncthreads();
    if (t < OPTIONS) {
        int c = 0;
        for (int r = 0; r < BATCH; ++r) c += (sIdx[r] == t);
        sCount[t] = c;
    }
    __syncthreads();
    if (t == 0) {
        int run = 0;
        for (int o = 0; o < OPTIONS; ++o) {
            sOff[o] = run;
            int tiles = (sCount[o] + BM - 1) >> 6;
            for (int j = 0; j < tiles; ++j) tileOpt[(run >> 6) + j] = o;
            run += tiles << 6;
        }
        for (int tt = run >> 6; tt < MT; ++tt) tileOpt[tt] = -1;
    }
    __syncthreads();
    for (int r = t; r < BATCH; r += 256) {
        int o = sIdx[r];
        int rank = 0;
        for (int j = 0; j < r; ++j) rank += (sIdx[j] == o);
        perm[sOff[o] + rank] = r;
    }
}

// store a (2k x 4n) micro-block into swizzled [n][k] bf16 LDS tile
static __device__ __forceinline__ void stageB(__bf16* __restrict__ bl,
                                              const float4& r0, const float4& r1,
                                              int n, int k) {
    int sw   = (n >> 4) & 3;                                 // uniform over n..n+3
    int base = n * KPAD + (((k >> 3) ^ sw) << 3) + (k & 7);
    const float a0[4] = {r0.x, r0.y, r0.z, r0.w};
    const float a1[4] = {r1.x, r1.y, r1.z, r1.w};
    #pragma unroll
    for (int j = 0; j < 4; ++j) {
        v2bf p; p.x = (__bf16)a0[j]; p.y = (__bf16)a1[j];    // one v_cvt_pk_bf16_f32
        *(v2bf*)&bl[base + j * KPAD] = p;                    // one ds_store_b32 (pairs to 2addr)
    }
}

static __device__ __forceinline__ void stageA(__bf16* __restrict__ dst, const float4& a) {
    v2bf p0; p0.x = (__bf16)a.x; p0.y = (__bf16)a.y;
    v2bf p1; p1.x = (__bf16)a.z; p1.y = (__bf16)a.w;
    *(v2bf*)dst = p0; *(v2bf*)(dst + 2) = p1;
}

// ---------------- bf16 WMMA GEMM, 64x128 block tile, double-buffered LDS ----------------
__global__ __launch_bounds__(256) void idxmlp_gemm_bf16(
    const float* __restrict__ A, const float* __restrict__ Wbase,
    const float* __restrict__ bias, float* __restrict__ Out,
    const int* __restrict__ perm, const int* __restrict__ tileOpt,
    int K, int N, int lda, int ldout, int gatherA, int scatterOut)
{
    __shared__ __bf16 Alds[2][BM * KPAD];   // [buf][row][k]
    __shared__ __bf16 Blds[2][BN * KPAD];   // [buf][n][k], 16B-chunk XOR swizzled

    int mt = blockIdx.y;
    int opt = tileOpt[mt];
    if (opt < 0) return;                           // uniform early-exit, before any WMMA
    const float* __restrict__ Bm = Wbase + (size_t)opt * (size_t)K * (size_t)N;
    int n0 = blockIdx.x * BN;

    int tid  = threadIdx.x;
    int lane = tid & 31;
    int wave = tid >> 5;
    int wm   = wave & 3;                  // 4 x 16-row slices   (BM = 64)
    int wg   = wave >> 2;                 // 2 x 64-col groups   (BN = 128), 4 tiles each

    // ---- A staging coords: 2 float4/thread over 64x32 fp32 ----
    int e0    = tid * 4;
    int arow0 = e0 >> 5;                  // 0..31 (c=1 adds 32)
    int acol  = e0 & 31;
    long srcRow[2];
    for (int c = 0; c < 2; ++c) {
        int gr = mt * BM + arow0 + c * 32;
        srcRow[c] = gatherA ? (long)perm[gr] : (long)gr;
    }
    // ---- B staging coords: 2 micro-blocks (2k x 4n) per thread over 32x128 fp32 ----
    int bk[2], bn[2];
    for (int c = 0; c < 2; ++c) {
        int e = tid + c * 256;
        bk[c] = (e >> 5) * 2;             // 0..30 even
        bn[c] = (e & 31) * 4;             // 0..124
    }

    v8f acc[4] = {{}, {}, {}, {}};
    int fr = lane & 15;
    int hi = lane >> 4;

    float4 aReg[2], bR0[2], bR1[2];
    // ---- prologue: stage k-tile 0 into buffer 0 ----
    for (int c = 0; c < 2; ++c) {
        aReg[c] = make_float4(0.f, 0.f, 0.f, 0.f);
        if (srcRow[c] >= 0)
            aReg[c] = *(const float4*)(A + srcRow[c] * (long)lda + acol);
        stageA(&Alds[0][(arow0 + c * 32) * KPAD + acol], aReg[c]);
    }
    for (int c = 0; c < 2; ++c) {
        const float* p = Bm + (size_t)bk[c] * N + n0 + bn[c];
        bR0[c] = *(const float4*)p;
        bR1[c] = *(const float4*)(p + N);
        stageB(&Blds[0][0], bR0[c], bR1[c], bn[c], bk[c]);
    }
    __syncthreads();

    int buf = 0;
    for (int k0 = 0; k0 < K; k0 += BK, buf ^= 1) {
        int kNext = k0 + BK;
        bool hasNext = kNext < K;

        // ---- issue next k-tile's global loads early (latency overlap with WMMA) ----
        if (hasNext) {
            for (int c = 0; c < 2; ++c) {
                aReg[c] = make_float4(0.f, 0.f, 0.f, 0.f);
                if (srcRow[c] >= 0)
                    aReg[c] = *(const float4*)(A + srcRow[c] * (long)lda + kNext + acol);
            }
            for (int c = 0; c < 2; ++c) {
                const float* p = Bm + (size_t)(kNext + bk[c]) * N + n0 + bn[c];
                if (kNext + BK < K)   // prefetch k-tile after next -> global_prefetch_b8
                    __builtin_prefetch(p + (size_t)BK * N, 0, 1);
                bR0[c] = *(const float4*)p;
                bR1[c] = *(const float4*)(p + N);
            }
        }

        // ---- load ALL fragments first (distinct regs -> batched ds_loads, partial waits) ----
        Frag fa;
        const __bf16* ap = &Alds[buf][(wm * 16 + fr) * KPAD + hi * 8];
        fa.h[0] = *(const v8bf*)ap;            // k = (lane>>4)*8 + 0..7
        fa.h[1] = *(const v8bf*)(ap + 16);     // k = (lane>>4)*8 + 16..23
        Frag fb[4];
        #pragma unroll
        for (int t = 0; t < 4; ++t) {
            int nn  = wg * 64 + t * 16 + fr;
            int sw2 = (wg * 4 + t) & 3;        // (nn>>4)&3, uniform over fr
            const __bf16* bbase = &Blds[buf][nn * KPAD];
            fb[t].h[0] = *(const v8bf*)(bbase + ((((hi     )    ) ^ sw2) << 3)); // k=hi*8+0..7
            fb[t].h[1] = *(const v8bf*)(bbase + ((((hi     ) | 2) ^ sw2) << 3)); // k=hi*8+16..23
        }
        // ---- back-to-back WMMAs (shared A, independent accumulators) ----
        #pragma unroll
        for (int t = 0; t < 4; ++t)
            acc[t] = __builtin_amdgcn_wmma_f32_16x16x32_bf16(false, fa.v, false, fb[t].v,
                                                             (short)0, acc[t], false, false);

        // ---- store next tile into alternate buffer ----
        if (hasNext) {
            for (int c = 0; c < 2; ++c)
                stageA(&Alds[buf ^ 1][(arow0 + c * 32) * KPAD + acol], aReg[c]);
            for (int c = 0; c < 2; ++c)
                stageB(&Blds[buf ^ 1][0], bR0[c], bR1[c], bn[c], bk[c]);
        }
        __syncthreads();
    }

    // ---- epilogue: + bias (option 0), optional scatter to original rows ----
    #pragma unroll
    for (int t = 0; t < 4; ++t) {
        int colg = n0 + wg * 64 + t * 16 + fr;
        float bv = bias[colg];
        #pragma unroll
        for (int r = 0; r < 8; ++r) {
            int gRow = mt * BM + wm * 16 + hi * 8 + r;   // C layout: row = (lane>>4)*8 + vgpr
            long orow = scatterOut ? (long)perm[gRow] : (long)gRow;
            if (orow >= 0) Out[orow * (long)ldout + colg] = acc[t][r] + bv;
        }
    }
}

extern "C" void kernel_launch(void* const* d_in, const int* in_sizes, int n_in,
                              void* d_out, int out_size, void* d_ws, size_t ws_size,
                              hipStream_t stream) {
    const float* x   = (const float*)d_in[0];
    const int*   idx = (const int*)  d_in[1];
    const float* W1  = (const float*)d_in[2];
    const float* b1  = (const float*)d_in[3];
    const float* W2  = (const float*)d_in[4];
    const float* b2  = (const float*)d_in[5];
    float* out = (float*)d_out;

    int*   perm    = (int*)d_ws;                         // PB ints
    int*   tileOpt = perm + PB;                          // MT ints
    float* hbuf    = (float*)((char*)d_ws + 16384);      // [PB x L1] sorted-order activations

    idxmlp_prep<<<1, 256, 0, stream>>>(idx, perm, tileOpt);

    // layer 1: gather x rows via perm -> hbuf (sorted, padded)
    idxmlp_gemm_bf16<<<dim3(L1 / BN, MT), 256, 0, stream>>>(
        x, W1, b1, hbuf, perm, tileOpt,
        /*K=*/L0, /*N=*/L1, /*lda=*/L0, /*ldout=*/L1, /*gatherA=*/1, /*scatterOut=*/0);

    // layer 2: read hbuf directly (sorted), scatter rows back via perm -> out
    idxmlp_gemm_bf16<<<dim3(L2 / BN, MT), 256, 0, stream>>>(
        hbuf, W2, b2, out, perm, tileOpt,
        /*K=*/L1, /*N=*/L2, /*lda=*/L1, /*ldout=*/L2, /*gatherA=*/0, /*scatterOut=*/1);
}
